// GlobalLossD_19636590477475
// MI455X (gfx1250) — matrix operand order repaired and checked by reading, not compiled, not run on hardware
//
#include <hip/hip_runtime.h>
#include <hip/hip_bf16.h>
#include <math.h>

// ---------------------------------------------------------------------------
// Problem constants (from reference): BATCH_SIZE=512 -> bs=2048, D=4096,
// N_PARTS=4, TEMP_FAC=0.1, active rows = 684, mask span = 160 offsets.
// ---------------------------------------------------------------------------
#define BS      2048
#define DIM     4096
#define NACT    684
#define SQRT10  3.16227766016837933f

typedef __attribute__((ext_vector_type(16))) __bf16       v16bf;
typedef __attribute__((ext_vector_type(8)))  float        v8f;
typedef __attribute__((ext_vector_type(4)))  unsigned int u32x4;

union FragU { v16bf v; u32x4 u[2]; };

// ---------------------------------------------------------------------------
// Kernel 1: row-normalize to bf16, folding sqrt(10) so Xn@Xn^T == S directly.
// ---------------------------------------------------------------------------
__global__ __launch_bounds__(256) void normalize_k(const float* __restrict__ X,
                                                   __bf16* __restrict__ Xn) {
  const int row = blockIdx.x;
  const float* xr = X + (size_t)row * DIM;
  float ss = 0.0f;
  for (int i = threadIdx.x; i < DIM; i += 256) {
    float v = xr[i];
    ss += v * v;
  }
  __shared__ float red[256];
  red[threadIdx.x] = ss;
  __syncthreads();
  for (int s = 128; s > 0; s >>= 1) {
    if (threadIdx.x < s) red[threadIdx.x] += red[threadIdx.x + s];
    __syncthreads();
  }
  const float scale = SQRT10 / fmaxf(sqrtf(red[0]), 1e-20f);
  __bf16* yr = Xn + (size_t)row * DIM;
  for (int i = threadIdx.x; i < DIM; i += 256)
    yr[i] = (__bf16)(xr[i] * scale);
}

// ---------------------------------------------------------------------------
// Kernel 2: S = Xn @ Xn^T via V_WMMA_F32_16X16X32_BF16.
// Block = 256 threads = 8 waves; block tile 128x128; wave tile 32x64
// (2x4 subtiles of 16x16). K-loop step 32 (bf16 WMMA K).
//
// A fragment (16x32 bf16, ISA layout): lane l = half*16+lm holds row lm,
//   elems 0..7  -> K = kk + 8*half .. +7      (16 B)
//   elems 8..15 -> K = kk + 16 + 8*half .. +7 (16 B)
// B fragment (32x16): column n = lane%16 of B == row n of Xn (B = Xn^T);
//   same K layout as A: 16 B at kk + 8*half and 16 B at kk + 16 + 8*half.
// ---------------------------------------------------------------------------
__global__ __launch_bounds__(256) void gemm_wmma(const __bf16* __restrict__ Xn,
                                                 float* __restrict__ S) {
  const int lane = threadIdx.x & 31;
  const int wid  = threadIdx.x >> 5;
  const int wm   = wid & 3;   // wave row group (0..3)
  const int wn   = wid >> 2;  // wave col group (0..1)
  const int lm   = lane & 15;
  const int half = lane >> 4;

  const int rowBase = blockIdx.y * 128 + wm * 32;
  const int colBase = blockIdx.x * 128 + wn * 64;

  const __bf16* aptr[2];
  const __bf16* bptr[4];
#pragma unroll
  for (int mi = 0; mi < 2; ++mi)
    aptr[mi] = Xn + (size_t)(rowBase + mi * 16 + lm) * DIM + half * 8;
#pragma unroll
  for (int ni = 0; ni < 4; ++ni)
    bptr[ni] = Xn + (size_t)(colBase + ni * 16 + lm) * DIM + half * 8;

  v8f acc[2][4];
#pragma unroll
  for (int mi = 0; mi < 2; ++mi)
#pragma unroll
    for (int ni = 0; ni < 4; ++ni) {
      v8f z = {0.f, 0.f, 0.f, 0.f, 0.f, 0.f, 0.f, 0.f};
      acc[mi][ni] = z;
    }

  for (int kk = 0; kk < DIM; kk += 32) {
    FragU a[2], b[4];
#pragma unroll
    for (int mi = 0; mi < 2; ++mi) {
      a[mi].u[0] = *(const u32x4*)(aptr[mi] + kk);
      a[mi].u[1] = *(const u32x4*)(aptr[mi] + kk + 16);
    }
#pragma unroll
    for (int ni = 0; ni < 4; ++ni) {
      b[ni].u[0] = *(const u32x4*)(bptr[ni] + kk);
      b[ni].u[1] = *(const u32x4*)(bptr[ni] + kk + 16);
    }
#pragma unroll
    for (int mi = 0; mi < 2; ++mi)
#pragma unroll
      for (int ni = 0; ni < 4; ++ni)
        acc[mi][ni] = __builtin_amdgcn_wmma_f32_16x16x32_bf16(
            false, a[mi].v, false, b[ni].v, (short)0, acc[mi][ni], false, false);
  }

  // C/D layout: VGPR g, lane l -> row = g + 8*(l/16), col = l%16.
#pragma unroll
  for (int mi = 0; mi < 2; ++mi) {
    const int r0 = rowBase + mi * 16 + half * 8;
#pragma unroll
    for (int ni = 0; ni < 4; ++ni) {
      const int c = colBase + ni * 16 + lm;
      float* outp = S + (size_t)r0 * BS + c;
#pragma unroll
      for (int g = 0; g < 8; ++g)
        outp[(size_t)g * BS] = acc[mi][ni][g];
    }
  }
}

// ---------------------------------------------------------------------------
// Kernel 3: RowSum[r] = sum_b exp(S[r,b])   (full, unmasked row sums)
// ---------------------------------------------------------------------------
__global__ __launch_bounds__(256) void rowsum_k(const float* __restrict__ S,
                                                float* __restrict__ RowSum) {
  const int r = blockIdx.x;
  const float* sr = S + (size_t)r * BS;
  float acc = 0.0f;
  for (int i = threadIdx.x; i < BS; i += 256)
    acc += expf(sr[i]);
  __shared__ float red[256];
  red[threadIdx.x] = acc;
  __syncthreads();
  for (int s = 128; s > 0; s >>= 1) {
    if (threadIdx.x < s) red[threadIdx.x] += red[threadIdx.x + s];
    __syncthreads();
  }
  if (threadIdx.x == 0) RowSum[r] = red[0];
}

// ---------------------------------------------------------------------------
// Kernel 4: Dsum[p,k] = RowSum[I[p,k]] - sum_{j<160} exp(S[I[p,k], (ap+4j)%bs])
// (M[p,:] has zeros exactly at offs = ap+4j mod bs, j=0..159).
// One wave per (p,k); 160 = 5*32 -> exactly 5 elements per lane.
// active[p] = 12*(p/4) + (p%4); I[p,k] = (active[p] + 4k) mod 2048.
// ---------------------------------------------------------------------------
__global__ __launch_bounds__(256) void dsum_k(const float* __restrict__ S,
                                              const float* __restrict__ RowSum,
                                              float* __restrict__ Dsum) {
  const int lane = threadIdx.x & 31;
  const int gid  = blockIdx.x * 8 + (threadIdx.x >> 5);
  if (gid >= NACT * 6) return;
  const int p  = gid / 6;
  const int k  = gid - p * 6;
  const int ap = 12 * (p >> 2) + (p & 3);
  const int r  = (ap + 4 * k) & (BS - 1);
  const float* sr = S + (size_t)r * BS;
  float partial = 0.0f;
#pragma unroll
  for (int t = 0; t < 5; ++t) {
    const int j = lane + t * 32;
    const int c = (ap + 4 * j) & (BS - 1);
    partial += expf(sr[c]);
  }
  for (int off = 16; off > 0; off >>= 1)
    partial += __shfl_xor(partial, off);
  if (lane == 0) Dsum[gid] = RowSum[r] - partial;
}

// ---------------------------------------------------------------------------
// Kernel 5: total = sum over p, (a,b) pairs of
//   log(en + Dsum[p,a]) + log(en + Dsum[p,b]) - 2*num,  en = exp(num),
//   num = S[I[p,a], I[p,b]];  output = total / 3072.
// Single block; deterministic LDS reduction.
// ---------------------------------------------------------------------------
__global__ __launch_bounds__(256) void loss_k(const float* __restrict__ S,
                                              const float* __restrict__ Dsum,
                                              float* __restrict__ out) {
  const int pa[7] = {0, 1, 0, 3, 4, 1, 2};
  const int pb[7] = {1, 2, 3, 4, 5, 4, 5};
  float acc = 0.0f;
  for (int idx = threadIdx.x; idx < NACT * 7; idx += 256) {
    const int p  = idx / 7;
    const int q  = idx - p * 7;
    const int ap = 12 * (p >> 2) + (p & 3);
    const int a  = pa[q], b = pb[q];
    const int ia = (ap + 4 * a) & (BS - 1);
    const int ib = (ap + 4 * b) & (BS - 1);
    const float num = S[(size_t)ia * BS + ib];
    const float en  = expf(num);
    acc += logf(en + Dsum[p * 6 + a]) + logf(en + Dsum[p * 6 + b]) - 2.0f * num;
  }
  __shared__ float red[256];
  red[threadIdx.x] = acc;
  __syncthreads();
  for (int s = 128; s > 0; s >>= 1) {
    if (threadIdx.x < s) red[threadIdx.x] += red[threadIdx.x + s];
    __syncthreads();
  }
  if (threadIdx.x == 0) out[0] = red[0] * (1.0f / 3072.0f);
}

// ---------------------------------------------------------------------------
// Launcher. Workspace layout (bytes):
//   [0, 16MB)          Xn   bf16 2048x4096
//   [16MB, 32MB)       S    f32  2048x2048
//   [32MB, 32MB+8KB)   RowSum f32 2048
//   [32MB+8KB, +16.4K) Dsum f32 684*6
// ---------------------------------------------------------------------------
extern "C" void kernel_launch(void* const* d_in, const int* in_sizes, int n_in,
                              void* d_out, int out_size, void* d_ws, size_t ws_size,
                              hipStream_t stream) {
  (void)in_sizes; (void)n_in; (void)out_size; (void)ws_size;
  const float* X = (const float*)d_in[0];
  float* out = (float*)d_out;

  char* ws = (char*)d_ws;
  __bf16* Xn     = (__bf16*)ws;
  float*  S      = (float*)(ws + (size_t)16 * 1024 * 1024);
  float*  RowSum = (float*)(ws + (size_t)32 * 1024 * 1024);
  float*  Dsum   = (float*)(ws + (size_t)32 * 1024 * 1024 + 8192);

  normalize_k<<<dim3(BS), dim3(256), 0, stream>>>(X, Xn);
  gemm_wmma<<<dim3(16, 16), dim3(256), 0, stream>>>(Xn, S);
  rowsum_k<<<dim3(BS), dim3(256), 0, stream>>>(S, RowSum);
  dsum_k<<<dim3((NACT * 6 + 7) / 8), dim3(256), 0, stream>>>(S, RowSum, Dsum);
  loss_k<<<dim3(1), dim3(256), 0, stream>>>(S, Dsum, out);
}
